// GCN_87170656239672
// MI455X (gfx1250) — matrix operand order repaired and checked by reading, not compiled, not run on hardware
//
#include <hip/hip_runtime.h>

#define NDIM 2048
#define BLK  128
#define KT   32
#define KITERS (NDIM / KT)

typedef __bf16 v16bf __attribute__((ext_vector_type(16)));
typedef __bf16 v8bf  __attribute__((ext_vector_type(8)));
typedef float  v8f   __attribute__((ext_vector_type(8)));
typedef int    v4i   __attribute__((ext_vector_type(4)));
typedef unsigned short us4v __attribute__((ext_vector_type(4)));
typedef unsigned short us8v __attribute__((ext_vector_type(8)));

typedef __attribute__((address_space(1))) v4i* gv4p; // global (HIP "__device__") ptr
typedef __attribute__((address_space(3))) v4i* lv4p; // LDS ("__shared__") ptr

#if __has_builtin(__builtin_amdgcn_global_load_async_to_lds_b128)
#define HAS_ASYNC_LDS 1
#else
#define HAS_ASYNC_LDS 0
#endif

// f32 -> bf16, round-to-nearest-even
static __device__ __forceinline__ unsigned short f2bf(float f) {
    union { float f; unsigned int u; } c; c.f = f;
    unsigned int u = c.u;
    u += 0x7FFFu + ((u >> 16) & 1u);
    return (unsigned short)(u >> 16);
}

static __device__ __forceinline__ void wait_async0() {
#if HAS_ASYNC_LDS
#if __has_builtin(__builtin_amdgcn_s_wait_asynccnt)
    __builtin_amdgcn_s_wait_asynccnt(0);
#else
    asm volatile("s_wait_asynccnt 0x0" ::: "memory");
#endif
#endif
}

// Copy 16 bytes global -> LDS. Async (ASYNCcnt-tracked, bypasses VGPRs) when
// the toolchain exposes the gfx1250 builtin; otherwise load+ds_store_b128.
static __device__ __forceinline__ void cp16(const unsigned short* g, unsigned short* l) {
#if HAS_ASYNC_LDS
    __builtin_amdgcn_global_load_async_to_lds_b128((gv4p)g, (lv4p)l, 0, 0);
#else
    *(v8bf*)l = *(const v8bf*)g;
#endif
}

// Assemble a 16-element bf16 fragment from two contiguous 16-byte LDS runs.
static __device__ __forceinline__ v16bf frag16(const unsigned short* p0,
                                               const unsigned short* p1) {
    v8bf lo = *(const v8bf*)p0;
    v8bf hi = *(const v8bf*)p1;
    v16bf r;
#pragma unroll
    for (int i = 0; i < 8; ++i) { r[i] = lo[i]; r[i + 8] = hi[i]; }
    return r;
}

// C = A*B (BT=false: B is bf16 [K][N]) or C = A*B^T (BT=true: B is bf16 [N][K]).
// A is bf16 [M][K]. fp32 accumulation. OUTBF selects bf16 vs f32 C.
// 128x128 block tile, 8 waves, 32x64 per wave, double-buffered LDS.
template <bool BT, bool OUTBF>
__global__ __launch_bounds__(256)
void gemm_wmma_bf16(const unsigned short* __restrict__ A,
                    const unsigned short* __restrict__ B, void* __restrict__ Cv) {
    __shared__ __align__(16) unsigned short As[2][BLK * KT]; // [m][k]
    __shared__ __align__(16) unsigned short Bs[2][BLK * KT]; // [n][k]

    const int tid  = threadIdx.x;
    const int lane = tid & 31;
    const int wave = tid >> 5;
    const int l16  = lane & 15;
    const int half = lane >> 4;
    const int mBase = (wave & 3) * 32;
    const int nBase = (wave >> 2) * 64;
    const int blockM = blockIdx.y * BLK;
    const int blockN = blockIdx.x * BLK;

    // per-thread staging chunks: 512 16B-chunks per tile, 2 per thread
    const int c0 = tid, c1 = tid + 256;

    v8f acc[2][4];
#pragma unroll
    for (int mt = 0; mt < 2; ++mt)
#pragma unroll
        for (int nt = 0; nt < 4; ++nt)
#pragma unroll
            for (int i = 0; i < 8; ++i) acc[mt][nt][i] = 0.0f;

    auto stage = [&](int k0, int nb) {
        // A tile [128 m][32 k]: straight 16B copies
#pragma unroll
        for (int p = 0; p < 2; ++p) {
            int c = p ? c1 : c0;
            int row = c >> 2, cv = c & 3;
            cp16(A + (size_t)(blockM + row) * NDIM + k0 + cv * 8,
                 &As[nb][row * KT + cv * 8]);
        }
        if (BT) {
            // B [N][K] row-major: same pattern as A
#pragma unroll
            for (int p = 0; p < 2; ++p) {
                int c = p ? c1 : c0;
                int row = c >> 2, cv = c & 3;
                cp16(B + (size_t)(blockN + row) * NDIM + k0 + cv * 8,
                     &Bs[nb][row * KT + cv * 8]);
            }
        } else {
            // B [K][N] row-major: transpose-scatter into LDS [n][k]
#pragma unroll
            for (int p = 0; p < 2; ++p) {
                int c = p ? c1 : c0;
                int kr = c >> 4, nv = c & 15;
                us8v v = *(const us8v*)(B + (size_t)(k0 + kr) * NDIM + blockN + nv * 8);
#pragma unroll
                for (int i = 0; i < 8; ++i)
                    Bs[nb][(nv * 8 + i) * KT + kr] = v[i];
            }
        }
    };

    stage(0, 0);
    wait_async0();
    __syncthreads();

    for (int kt = 0; kt < KITERS; ++kt) {
        const int buf = kt & 1;
        if (kt + 1 < KITERS) stage((kt + 1) * KT, 1 - buf);

        const unsigned short* Ab = As[buf];
        const unsigned short* Bb = Bs[buf];
        v16bf afrag[2];
#pragma unroll
        for (int mt = 0; mt < 2; ++mt) {
            const unsigned short* ar = &Ab[(mBase + mt * 16 + l16) * KT];
            afrag[mt] = frag16(ar + 8 * half, ar + 16 + 8 * half);
        }
#pragma unroll
        for (int nt = 0; nt < 4; ++nt) {
            const unsigned short* br = &Bb[(nBase + nt * 16 + l16) * KT];
            v16bf bfrag = frag16(br + 8 * half, br + 16 + 8 * half);
#pragma unroll
            for (int mt = 0; mt < 2; ++mt) {
                acc[mt][nt] = __builtin_amdgcn_wmma_f32_16x16x32_bf16(
                    false, afrag[mt], false, bfrag, (short)0, acc[mt][nt], false, false);
            }
        }
        if (kt + 1 < KITERS) wait_async0();
        __syncthreads();
    }

    // epilogue: C/D layout VGPR i -> M = 8*half + i, N = lane%16
#pragma unroll
    for (int mt = 0; mt < 2; ++mt) {
#pragma unroll
        for (int nt = 0; nt < 4; ++nt) {
            int gn  = blockN + nBase + nt * 16 + l16;
            int gm0 = blockM + mBase + mt * 16 + 8 * half;
#pragma unroll
            for (int i = 0; i < 8; ++i) {
                size_t off = (size_t)(gm0 + i) * NDIM + gn;
                if (OUTBF) ((unsigned short*)Cv)[off] = f2bf(acc[mt][nt][i]);
                else       ((float*)Cv)[off] = acc[mt][nt][i];
            }
        }
    }
}

// dst_bf16[i] = bf16(src[i]), vectorized x4
__global__ __launch_bounds__(256)
void convert_bf16_kernel(const float* __restrict__ src, unsigned short* __restrict__ dst,
                         int n4) {
    int i = blockIdx.x * 256 + threadIdx.x;
    if (i < n4) {
        float4 v = ((const float4*)src)[i];
        us4v h;
        h[0] = f2bf(v.x); h[1] = f2bf(v.y); h[2] = f2bf(v.z); h[3] = f2bf(v.w);
        ((us4v*)dst)[i] = h;
    }
}

// r[i] = sum_j (h[i,j]*h[j,i])^2
__global__ __launch_bounds__(256)
void rowsum_sq_kernel(const float* __restrict__ h, float* __restrict__ r) {
    __shared__ float red[256];
    const int i = blockIdx.x, tid = threadIdx.x;
    float s = 0.f;
    for (int j = tid; j < NDIM; j += 256) {
        float t = h[(size_t)i * NDIM + j] * h[(size_t)j * NDIM + i];
        s += t * t;
    }
    red[tid] = s; __syncthreads();
    for (int off = 128; off > 0; off >>= 1) {
        if (tid < off) red[tid] += red[tid + off];
        __syncthreads();
    }
    if (tid == 0) r[i] = red[0];
}

// lap[i,j] = 0.5*((h[i,j]*h[j,i])^2 / r[i] + (i==j)), written directly as bf16
// (adj rows sum to 1 => degree == 2 => s_i = rsqrt(2) for all i)
__global__ __launch_bounds__(256)
void build_lap_kernel(const float* __restrict__ h, const float* __restrict__ r,
                      unsigned short* __restrict__ lap) {
    const int i = blockIdx.x, tid = threadIdx.x;
    const float inv = 0.5f / r[i];
    for (int j = tid; j < NDIM; j += 256) {
        float t = h[(size_t)i * NDIM + j] * h[(size_t)j * NDIM + i];
        float v = t * t * inv;
        if (j == i) v += 0.5f;
        lap[(size_t)i * NDIM + j] = f2bf(v);
    }
}

// LayerNorm over last dim + LeakyReLU(0.1).
// LAST: write f32 final output; else write bf16 activations for the next GEMM.
template <bool LAST>
__global__ __launch_bounds__(256)
void layernorm_leaky_kernel(const float* __restrict__ y, const float* __restrict__ g,
                            const float* __restrict__ b, void* __restrict__ dstv) {
    __shared__ float r1[256], r2[256];
    const int i = blockIdx.x, tid = threadIdx.x;
    float s1 = 0.f, s2 = 0.f;
    for (int j = tid; j < NDIM; j += 256) {
        float v = y[(size_t)i * NDIM + j];
        s1 += v; s2 += v * v;
    }
    r1[tid] = s1; r2[tid] = s2; __syncthreads();
    for (int off = 128; off > 0; off >>= 1) {
        if (tid < off) { r1[tid] += r1[tid + off]; r2[tid] += r2[tid + off]; }
        __syncthreads();
    }
    const float mu  = r1[0] * (1.0f / NDIM);
    const float var = r2[0] * (1.0f / NDIM) - mu * mu;
    const float inv = rsqrtf(var + 1e-5f);
    for (int j = tid; j < NDIM; j += 256) {
        float v = (y[(size_t)i * NDIM + j] - mu) * inv * g[j] + b[j];
        v = v >= 0.f ? v : 0.1f * v;
        if (LAST) ((float*)dstv)[(size_t)i * NDIM + j] = v;
        else      ((unsigned short*)dstv)[(size_t)i * NDIM + j] = f2bf(v);
    }
}

extern "C" void kernel_launch(void* const* d_in, const int* in_sizes, int n_in,
                              void* d_out, int out_size, void* d_ws, size_t ws_size,
                              hipStream_t stream) {
    const float* x     = (const float*)d_in[0];
    const float* W1    = (const float*)d_in[1];
    const float* Ws    = (const float*)d_in[2];
    const float* gamma = (const float*)d_in[3];
    const float* beta  = (const float*)d_in[4];

    const size_t NN = (size_t)NDIM * NDIM;

    // workspace layout: f32 h first, then bf16 pools (~88 MB total)
    float*          h     = (float*)d_ws;                // NN f32   (16 MB)
    float*          r     = h + NN;                      // NDIM f32
    unsigned short* xbf   = (unsigned short*)(r + NDIM); // NN bf16  (8 MB)
    unsigned short* w1bf  = xbf  + NN;                   // NN bf16
    unsigned short* wsbf  = w1bf + NN;                   // 4*NN bf16 (32 MB)
    unsigned short* lapbf = wsbf + 4 * NN;               // NN bf16
    unsigned short* tmpbf = lapbf + NN;                  // NN bf16
    unsigned short* curbf = tmpbf + NN;                  // NN bf16

    dim3 ggrid(NDIM / BLK, NDIM / BLK);
    const int n4 = (int)(NN / 4);

    // one-time operand conversion to bf16
    convert_bf16_kernel<<<(n4 + 255) / 256, 256, 0, stream>>>(x, xbf, n4);
    convert_bf16_kernel<<<(n4 + 255) / 256, 256, 0, stream>>>(W1, w1bf, n4);
    convert_bf16_kernel<<<(4 * n4 + 255) / 256, 256, 0, stream>>>(Ws, wsbf, 4 * n4);

    // h = x @ W1^T  (f32 out: feeds laplacian construction)
    gemm_wmma_bf16<true, false><<<ggrid, 256, 0, stream>>>(xbf, w1bf, h);
    rowsum_sq_kernel<<<NDIM, 256, 0, stream>>>(h, r);
    build_lap_kernel<<<NDIM, 256, 0, stream>>>(h, r, lapbf);

    const unsigned short* src = xbf;
    for (int j = 0; j < 4; ++j) {
        // tmp = lap @ src      (bf16 out: only consumed as GEMM A operand)
        gemm_wmma_bf16<false, true><<<ggrid, 256, 0, stream>>>(lapbf, src, tmpbf);
        // h = tmp @ Ws[j]^T    (f32 out: feeds LayerNorm)
        gemm_wmma_bf16<true, false><<<ggrid, 256, 0, stream>>>(tmpbf, wsbf + (size_t)j * NN, h);
        if (j == 3)
            layernorm_leaky_kernel<true><<<NDIM, 256, 0, stream>>>(
                h, gamma + (size_t)j * NDIM, beta + (size_t)j * NDIM, d_out);
        else
            layernorm_leaky_kernel<false><<<NDIM, 256, 0, stream>>>(
                h, gamma + (size_t)j * NDIM, beta + (size_t)j * NDIM, curbf);
        src = curbf;
    }

    (void)in_sizes; (void)n_in; (void)out_size; (void)ws_size;
}
